// ViT_9869834846804
// MI455X (gfx1250) — compile-verified, matrix operand order checked by reading, not run on hardware
//
#include <hip/hip_runtime.h>
#include <hip/hip_bf16.h>
#include <math.h>

// ---------------------------------------------------------------------------
// DiT-ViT forward for MI455X (gfx1250, wave32, WMMA, async-LDS).
// bf16 end-to-end operands (weights pre-transposed/converted once, activations
// produced in bf16), f32 WMMA accumulation, fp32 residual stream.
// ---------------------------------------------------------------------------

typedef __attribute__((ext_vector_type(16))) __bf16 v16bf;
typedef __attribute__((ext_vector_type(8)))  float  v8f;
typedef __attribute__((ext_vector_type(4)))  int    v4i;

#define T_TOK   1024
#define DMODEL  384
#define DFF     1536
#define NHEAD   8
#define DHEAD   48
#define NROWS   8192      // B * T_TOK
#define PDIM    768       // 3*16*16

// ---------------------------------------------------------------------------
// gfx1250 async global->LDS copy (ASYNCcnt path). 16 bytes per lane.
// Builtin signature (from clang diagnostic): (global v4i* src, local v4i* dst,
// imm offset, imm cpol). Falls back to vector load + ds store if unavailable.
// ---------------------------------------------------------------------------
#if defined(__gfx1250__) && __has_builtin(__builtin_amdgcn_global_load_async_to_lds_b128)
#define HAVE_ASYNC_LDS 1
#else
#define HAVE_ASYNC_LDS 0
#endif

__device__ static inline void async_cp16(const void* g, void* l) {
#if HAVE_ASYNC_LDS
  typedef v4i __attribute__((address_space(1)))* gptr_t;
  typedef v4i __attribute__((address_space(3)))* lptr_t;
  __builtin_amdgcn_global_load_async_to_lds_b128((gptr_t)(g), (lptr_t)(l), 0, 0);
#else
  *(uint4*)l = *(const uint4*)g;
#endif
}

__device__ static inline void async_wait0() {
#if HAVE_ASYNC_LDS
#if __has_builtin(__builtin_amdgcn_s_wait_asynccnt)
  __builtin_amdgcn_s_wait_asynccnt(0);
#else
  asm volatile("s_wait_asynccnt 0" ::: "memory");
#endif
#endif
}

// ---------------------------------------------------------------------------
// Build a 16x32 bf16 WMMA operand fragment from LDS (ISA 7.12.2 layout):
// lane L(0..15) -> row L, K-half 0; lane 16+L -> row L, K-half 1 (kh=8).
// VGPR v<4 holds K = kh+2v(+1); VGPR 4+v holds K = 16+kh+2v(+1).
// ---------------------------------------------------------------------------
__device__ static inline v16bf load_frag(const __bf16* base, int stride,
                                         int lane, int kofs) {
  v16bf f;
  const int r  = lane & 15;
  const int kh = (lane >> 4) * 8;
  const __bf16* p = base + r * stride + kofs + kh;
#pragma unroll
  for (int v = 0; v < 4; ++v) {
    f[2 * v + 0] = p[2 * v + 0];
    f[2 * v + 1] = p[2 * v + 1];
    f[2 * v + 8] = p[16 + 2 * v + 0];
    f[2 * v + 9] = p[16 + 2 * v + 1];
  }
  return f;
}

// ---------------------------------------------------------------------------
// WMMA GEMM: C[M,N] = epi(A[M,K] @ W[K,N] + bias[N]), A bf16 row-major,
// W supplied pre-transposed bf16 Wt[N][K]. Block tile 128x64, K-step 32,
// 256 threads / 8 waves; each wave does a 32x32 subtile = 4 WMMAs per K-step.
// Epilogue: RELU / RES (C += res fp32) / POS (C += pos[row%1024]) and fp32 or
// bf16 output. Tiles staged via async global->LDS copies (ASYNCcnt).
// ---------------------------------------------------------------------------
#define LDT 40  // padded LDS row stride in bf16 (80 B, 16 B aligned)

template <bool RELU, bool RES, bool POS, bool OUTBF>
__global__ __launch_bounds__(256) void gemm_kernel(
    const __bf16* __restrict__ A, const __bf16* __restrict__ Wt,
    const float* __restrict__ bias, const float* __restrict__ res,
    const float* __restrict__ pos, void* __restrict__ C,
    int M, int N, int K) {
  __shared__ __bf16 a_lds[128][LDT];
  __shared__ __bf16 w_lds[64][LDT];

  const int tid  = threadIdx.x;
  const int lane = tid & 31;
  const int wave = tid >> 5;
  const int wm   = wave & 3;   // 4 row groups of 32
  const int wn   = wave >> 2;  // 2 col groups of 32
  const int row0 = blockIdx.y * 128;
  const int col0 = blockIdx.x * 64;

  // fill mapping: A 128x32 (16 bf16/thread = 2x16B), W 64x32 (8 bf16 = 1x16B)
  const int ar = tid >> 1, ah = (tid & 1) * 16;
  const int wr = tid >> 2, wq = (tid & 3) * 8;

  const bool a_inr = (row0 + ar) < M;   // ragged M (time-MLPs, M=8)
  if (!a_inr) {
    const uint4 z = {0u, 0u, 0u, 0u};
    *(uint4*)&a_lds[ar][ah]     = z;
    *(uint4*)&a_lds[ar][ah + 8] = z;
  }

  v8f acc00 = {}, acc01 = {}, acc10 = {}, acc11 = {};

  for (int k0 = 0; k0 < K; k0 += 32) {
    if (a_inr) {
      const __bf16* ap = A + (size_t)(row0 + ar) * K + k0 + ah;
      async_cp16(ap,     &a_lds[ar][ah]);
      async_cp16(ap + 8, &a_lds[ar][ah + 8]);
    }
    async_cp16(Wt + (size_t)(col0 + wr) * K + k0 + wq, &w_lds[wr][wq]);
    async_wait0();
    __syncthreads();

    v16bf af0 = load_frag(&a_lds[wm * 32][0],      LDT, lane, 0);
    v16bf af1 = load_frag(&a_lds[wm * 32 + 16][0], LDT, lane, 0);
    v16bf bf0 = load_frag(&w_lds[wn * 32][0],      LDT, lane, 0);
    v16bf bf1 = load_frag(&w_lds[wn * 32 + 16][0], LDT, lane, 0);
    acc00 = __builtin_amdgcn_wmma_f32_16x16x32_bf16(false, af0, false, bf0, (short)0, acc00, false, false);
    acc01 = __builtin_amdgcn_wmma_f32_16x16x32_bf16(false, af0, false, bf1, (short)0, acc01, false, false);
    acc10 = __builtin_amdgcn_wmma_f32_16x16x32_bf16(false, af1, false, bf0, (short)0, acc10, false, false);
    acc11 = __builtin_amdgcn_wmma_f32_16x16x32_bf16(false, af1, false, bf1, (short)0, acc11, false, false);
    __syncthreads();
  }

  const int lh = lane >> 4, lc = lane & 15;
#pragma unroll
  for (int rg = 0; rg < 2; ++rg) {
    const v8f* ar0 = rg ? &acc10 : &acc00;
    const v8f* ar1 = rg ? &acc11 : &acc01;
#pragma unroll
    for (int j = 0; j < 8; ++j) {
      const int row = row0 + wm * 32 + rg * 16 + j + 8 * lh;
      if (row < M) {
        const int c0 = col0 + wn * 32 + lc;
        const int c1 = c0 + 16;
        float v0 = (*ar0)[j] + bias[c0];
        float v1 = (*ar1)[j] + bias[c1];
        if (RELU) { v0 = fmaxf(v0, 0.0f); v1 = fmaxf(v1, 0.0f); }
        const size_t base = (size_t)row * N;
        if (RES) { v0 += res[base + c0]; v1 += res[base + c1]; }
        if (POS) {
          const size_t pb = (size_t)(row & (T_TOK - 1)) * N;
          v0 += pos[pb + c0]; v1 += pos[pb + c1];
        }
        if (OUTBF) {
          ((__bf16*)C)[base + c0] = (__bf16)v0;
          ((__bf16*)C)[base + c1] = (__bf16)v1;
        } else {
          ((float*)C)[base + c0] = v0;
          ((float*)C)[base + c1] = v1;
        }
      }
    }
  }
}

// ---------------------------------------------------------------------------
// Flash attention: grid.x = B*NHEAD*(T/16); 1 wave per 16-query tile.
// bf16 q/k/v in, bf16 out. dh=48 contracted as two K=32 WMMAs (zero-padded).
// Streaming softmax with the reference's 1/sqrt(384) scale.
// ---------------------------------------------------------------------------
__global__ __launch_bounds__(32) void attn_kernel(
    const __bf16* __restrict__ Q, const __bf16* __restrict__ K,
    const __bf16* __restrict__ V, __bf16* __restrict__ O) {
  __shared__ __bf16 q_lds[16][64];
  __shared__ __bf16 k_lds[32][64];
  __shared__ __bf16 vT_lds[48][32];
  __shared__ float  s_lds[16][32];
  __shared__ __bf16 p_lds[16][32];
  __shared__ float  corr_lds[16];
  __shared__ float  l_lds[16];

  const int lane = threadIdx.x;
  const int blk  = blockIdx.x;
  const int qb   = blk & 63;
  const int head = (blk >> 6) & 7;
  const int b    = blk >> 9;
  const int qrow0 = b * T_TOK + qb * 16;
  const int hcol  = head * DHEAD;
  const float scale = 0.05103103630798287f;  // 1/sqrt(384) (full-D, per ref)

  for (int i = lane; i < 16 * 16; i += 32) q_lds[i >> 4][48 + (i & 15)] = (__bf16)0.0f;
  for (int i = lane; i < 32 * 16; i += 32) k_lds[i >> 4][48 + (i & 15)] = (__bf16)0.0f;
  for (int i = lane; i < 16 * 48; i += 32) {
    const int r = i / 48, c = i % 48;
    q_lds[r][c] = Q[(size_t)(qrow0 + r) * DMODEL + hcol + c];
  }
  __syncthreads();

  const v16bf aq0 = load_frag(&q_lds[0][0], 64, lane, 0);
  const v16bf aq1 = load_frag(&q_lds[0][0], 64, lane, 32);

  v8f oacc0 = {}, oacc1 = {}, oacc2 = {};
  float m_run = -3.0e38f, l_run = 0.0f;  // lanes 0..15
  const int lh = lane >> 4, lc = lane & 15;

  for (int kc = 0; kc < T_TOK; kc += 32) {
    __syncthreads();
    {
      const int krow = b * T_TOK + kc + lane;  // lane owns one key row
      const __bf16* kp = K + (size_t)krow * DMODEL + hcol;
      const __bf16* vp = V + (size_t)krow * DMODEL + hcol;
      for (int c = 0; c < DHEAD; ++c) {
        k_lds[lane][c]  = kp[c];
        vT_lds[c][lane] = vp[c];
      }
    }
    __syncthreads();

#pragma unroll
    for (int s = 0; s < 2; ++s) {
      v16bf bk0 = load_frag(&k_lds[s * 16][0], 64, lane, 0);
      v16bf bk1 = load_frag(&k_lds[s * 16][0], 64, lane, 32);
      v8f sacc = {};
      sacc = __builtin_amdgcn_wmma_f32_16x16x32_bf16(false, aq0, false, bk0, (short)0, sacc, false, false);
      sacc = __builtin_amdgcn_wmma_f32_16x16x32_bf16(false, aq1, false, bk1, (short)0, sacc, false, false);
#pragma unroll
      for (int j = 0; j < 8; ++j) s_lds[j + 8 * lh][s * 16 + lc] = sacc[j];
    }
    __syncthreads();

    if (lane < 16) {
      float mc = -3.0e38f;
      for (int c = 0; c < 32; ++c) mc = fmaxf(mc, s_lds[lane][c] * scale);
      const float mnew = fmaxf(m_run, mc);
      const float corr = __expf(m_run - mnew);
      float ssum = 0.0f;
      for (int c = 0; c < 32; ++c) {
        const float p = __expf(s_lds[lane][c] * scale - mnew);
        p_lds[lane][c] = (__bf16)p;
        ssum += p;
      }
      l_run = l_run * corr + ssum;
      m_run = mnew;
      corr_lds[lane] = corr;
    }
    __syncthreads();

#pragma unroll
    for (int j = 0; j < 8; ++j) {
      const float f = corr_lds[j + 8 * lh];
      oacc0[j] *= f; oacc1[j] *= f; oacc2[j] *= f;
    }

    v16bf ap  = load_frag(&p_lds[0][0],   32, lane, 0);
    v16bf bv0 = load_frag(&vT_lds[0][0],  32, lane, 0);
    v16bf bv1 = load_frag(&vT_lds[16][0], 32, lane, 0);
    v16bf bv2 = load_frag(&vT_lds[32][0], 32, lane, 0);
    oacc0 = __builtin_amdgcn_wmma_f32_16x16x32_bf16(false, ap, false, bv0, (short)0, oacc0, false, false);
    oacc1 = __builtin_amdgcn_wmma_f32_16x16x32_bf16(false, ap, false, bv1, (short)0, oacc1, false, false);
    oacc2 = __builtin_amdgcn_wmma_f32_16x16x32_bf16(false, ap, false, bv2, (short)0, oacc2, false, false);
  }

  if (lane < 16) l_lds[lane] = l_run;
  __syncthreads();
#pragma unroll
  for (int j = 0; j < 8; ++j) {
    const int row = j + 8 * lh;
    const float inv = 1.0f / l_lds[row];
    __bf16* op = O + (size_t)(qrow0 + row) * DMODEL + hcol;
    op[lc]      = (__bf16)(oacc0[j] * inv);
    op[16 + lc] = (__bf16)(oacc1[j] * inv);
    op[32 + lc] = (__bf16)(oacc2[j] * inv);
  }
}

// ---------------------------------------------------------------------------
// LayerNorm over D=384 (one wave/row), fp32 in -> bf16 out, optional fp32
// per-batch time-vector broadcast add.
// ---------------------------------------------------------------------------
__global__ __launch_bounds__(256) void ln_kernel(
    const float* __restrict__ x, const float* __restrict__ g,
    const float* __restrict__ bb, const float* __restrict__ tvec,
    __bf16* __restrict__ out, int rows) {
  const int lane = threadIdx.x & 31;
  const int row  = blockIdx.x * 8 + (threadIdx.x >> 5);
  if (row >= rows) return;
  const float* xr = x + (size_t)row * DMODEL;
  float v[12];
  float s = 0.0f;
#pragma unroll
  for (int i = 0; i < 12; ++i) { v[i] = xr[lane + i * 32]; s += v[i]; }
#pragma unroll
  for (int o = 16; o > 0; o >>= 1) s += __shfl_xor(s, o, 32);
  const float mu = s * (1.0f / DMODEL);
  float var = 0.0f;
#pragma unroll
  for (int i = 0; i < 12; ++i) { const float d = v[i] - mu; var += d * d; }
#pragma unroll
  for (int o = 16; o > 0; o >>= 1) var += __shfl_xor(var, o, 32);
  const float r = rsqrtf(var * (1.0f / DMODEL) + 1e-5f);
  const int bidx = row >> 10;
  __bf16* orow = out + (size_t)row * DMODEL;
#pragma unroll
  for (int i = 0; i < 12; ++i) {
    const int c = lane + i * 32;
    float o = (v[i] - mu) * r * g[c] + bb[c];
    if (tvec) o += tvec[bidx * DMODEL + c];
    orow[c] = (__bf16)o;
  }
}

// --------------------------- weight conversion -----------------------------
// Per-layer: slots [q,k,v,o,t1a,t1b,t2a,t2b] 384x384, [f1] K=384,N=1536,
// [f2] K=1536,N=384. Writes transposed bf16 Wt[N][K] packed contiguously.
struct ConvArgs10 { const float* w[10]; };

__global__ __launch_bounds__(256) void convert_layer_kernel(ConvArgs10 a,
                                                            __bf16* __restrict__ dst) {
  const int i = blockIdx.x * 256 + threadIdx.x;
  if (i >= 2359296) return;
  int slot, r, K, N; size_t dofs;
  if (i < 1179648)      { slot = i / 147456; r = i % 147456; K = 384;  N = 384;  dofs = (size_t)slot * 147456; }
  else if (i < 1769472) { slot = 8; r = i - 1179648; K = 384;  N = 1536; dofs = 1179648; }
  else                  { slot = 9; r = i - 1769472; K = 1536; N = 384;  dofs = 1769472; }
  const int n = r / K, k = r % K;
  dst[dofs + r] = (__bf16)a.w[slot][(size_t)k * N + n];
}

__global__ __launch_bounds__(256) void convert_head_kernel(
    const float* __restrict__ pw, const float* __restrict__ ow,
    __bf16* __restrict__ pwt, __bf16* __restrict__ owt) {
  const int i = blockIdx.x * 256 + threadIdx.x;
  if (i >= 589824) return;
  if (i < 294912) { const int n = i / 768, k = i % 768; pwt[i] = (__bf16)pw[(size_t)k * 384 + n]; }
  else { const int r = i - 294912; const int n = r / 384, k = r % 384; owt[r] = (__bf16)ow[(size_t)k * 768 + n]; }
}

// --------------------------- index-remap kernels ---------------------------
__global__ void patchify_kernel(const float* __restrict__ x, __bf16* __restrict__ xp) {
  const int i = blockIdx.x * 256 + threadIdx.x;
  if (i >= NROWS * PDIM) return;
  const int row = i / PDIM, col = i % PDIM;
  const int b = row >> 10, tk = row & 1023, hb = tk >> 5, wb = tk & 31;
  const int c = col % 3, pp = col / 3, ph = pp >> 4, pw = pp & 15;
  xp[i] = (__bf16)x[((size_t)((b * 3 + c) * 512 + hb * 16 + ph)) * 512 + wb * 16 + pw];
}

__global__ void unpatchify_kernel(const float* __restrict__ o, float* __restrict__ out) {
  const int i = blockIdx.x * 256 + threadIdx.x;
  if (i >= 8 * 3 * 512 * 512) return;
  const int b = i / (3 * 512 * 512);
  const int r = i % (3 * 512 * 512);
  const int c = r / (512 * 512);
  const int hw = r % (512 * 512);
  const int h = hw >> 9, w = hw & 511;
  const int row = b * T_TOK + (h >> 4) * 32 + (w >> 4);
  const int col = ((h & 15) * 16 + (w & 15)) * 3 + c;
  out[i] = o[(size_t)row * PDIM + col];
}

__global__ void temb_gather_kernel(const int* __restrict__ t,
                                   const float* __restrict__ table,
                                   __bf16* __restrict__ temb) {
  const int i = blockIdx.x * 256 + threadIdx.x;
  if (i >= 8 * DMODEL) return;
  const int b = i / DMODEL, c = i % DMODEL;
  temb[i] = (__bf16)table[(size_t)t[b] * DMODEL + c];
}

// ------------------------------- host driver -------------------------------
extern "C" void kernel_launch(void* const* d_in, const int* in_sizes, int n_in,
                              void* d_out, int out_size, void* d_ws, size_t ws_size,
                              hipStream_t stream) {
  // setup_inputs() insertion order, recursively (w before b in each dense):
  const float* x          = (const float*)d_in[0];
  const int*   tt         = (const int*)d_in[1];
  const float* patch_w    = (const float*)d_in[2];
  const float* patch_b    = (const float*)d_in[3];
  const float* out_w      = (const float*)d_in[4];
  const float* out_b      = (const float*)d_in[5];
  const float* time_embed = (const float*)d_in[6];
  const float* pos_embed  = (const float*)d_in[7];
  const float* fln_g      = (const float*)d_in[8];
  const float* fln_b      = (const float*)d_in[9];
  const int LBASE = 10, LSTRIDE = 24;
  // per layer: q.w q.b k.w k.b v.w v.b o.w o.b f1.w f1.b f2.w f2.b
  //            t1a.w t1a.b t1b.w t1b.b t2a.w t2a.b t2b.w t2b.b
  //            ln1_g ln1_b ln2_g ln2_b

  // ---- workspace carving: bf16 region first (all counts multiples of 16) --
  const size_t LW = 2359296;  // bf16 weights per layer
  __bf16* wb = (__bf16*)d_ws;
  __bf16* lwt[8];
  for (int l = 0; l < 8; ++l) { lwt[l] = wb; wb += LW; }
  __bf16* patch_wt = wb; wb += 294912;
  __bf16* out_wt   = wb; wb += 294912;
  __bf16* xp_b  = wb; wb += (size_t)NROWS * PDIM;
  __bf16* xln_b = wb; wb += (size_t)NROWS * DMODEL;
  __bf16* qb    = wb; wb += (size_t)NROWS * DMODEL;
  __bf16* kb    = wb; wb += (size_t)NROWS * DMODEL;
  __bf16* vb    = wb; wb += (size_t)NROWS * DMODEL;
  __bf16* ab    = wb; wb += (size_t)NROWS * DMODEL;
  __bf16* ff_b  = wb; wb += (size_t)NROWS * DFF;
  __bf16* temb_b = wb; wb += 8 * DMODEL;
  __bf16* th_b   = wb; wb += 8 * DMODEL;
  float* fb = (float*)wb;
  float* h    = fb; fb += (size_t)NROWS * DMODEL;
  float* outf = fb; fb += (size_t)NROWS * PDIM;
  float* tv   = fb; fb += 8 * DMODEL;

  const dim3 gD(DMODEL / 64, NROWS / 128);  // 6 x 64
  const dim3 gF(DFF / 64, NROWS / 128);     // 24 x 64
  const dim3 gO(PDIM / 64, NROWS / 128);    // 12 x 64
  const dim3 gT(DMODEL / 64, 1);            // time-MLPs (M=8)

  // ---- one-time per-call weight convert+transpose to bf16 ----
  for (int l = 0; l < 8; ++l) {
    const float* const* L = (const float* const*)(d_in + LBASE + l * LSTRIDE);
    ConvArgs10 ca;
    ca.w[0] = L[0];  ca.w[1] = L[2];  ca.w[2] = L[4];  ca.w[3] = L[6];   // q k v o
    ca.w[4] = L[12]; ca.w[5] = L[14]; ca.w[6] = L[16]; ca.w[7] = L[18];  // t1a t1b t2a t2b
    ca.w[8] = L[8];  ca.w[9] = L[10];                                    // f1 f2
    convert_layer_kernel<<<(2359296 + 255) / 256, 256, 0, stream>>>(ca, lwt[l]);
  }
  convert_head_kernel<<<(589824 + 255) / 256, 256, 0, stream>>>(patch_w, out_w, patch_wt, out_wt);

  patchify_kernel<<<(NROWS * PDIM + 255) / 256, 256, 0, stream>>>(x, xp_b);
  temb_gather_kernel<<<(8 * DMODEL + 255) / 256, 256, 0, stream>>>(tt, time_embed, temb_b);

  // h = patch(xp) + pos_embed[token]   (fp32 residual stream)
  gemm_kernel<false, false, true, false><<<gD, 256, 0, stream>>>(
      xp_b, patch_wt, patch_b, nullptr, pos_embed, h, NROWS, DMODEL, PDIM);

  for (int l = 0; l < 8; ++l) {
    const float* const* L = (const float* const*)(d_in + LBASE + l * LSTRIDE);
    const float *qbs = L[1], *kbs = L[3], *vbs = L[5], *obs = L[7];
    const float *f1b = L[9], *f2b = L[11];
    const float *t1ab = L[13], *t1bb = L[15], *t2ab = L[17], *t2bb = L[19];
    const float *ln1g = L[20], *ln1b = L[21], *ln2g = L[22], *ln2b = L[23];
    __bf16* W = lwt[l];
    __bf16 *Wq = W, *Wk = W + 147456, *Wv = W + 2 * 147456, *Wo = W + 3 * 147456;
    __bf16 *Wt1a = W + 4 * 147456, *Wt1b = W + 5 * 147456;
    __bf16 *Wt2a = W + 6 * 147456, *Wt2b = W + 7 * 147456;
    __bf16 *Wf1 = W + 1179648, *Wf2 = W + 1769472;

    // tvec1 = t1b(relu(t1a(temb)))
    gemm_kernel<true,  false, false, true ><<<gT, 256, 0, stream>>>(temb_b, Wt1a, t1ab, nullptr, nullptr, th_b, 8, DMODEL, DMODEL);
    gemm_kernel<false, false, false, false><<<gT, 256, 0, stream>>>(th_b,   Wt1b, t1bb, nullptr, nullptr, tv,   8, DMODEL, DMODEL);
    ln_kernel<<<NROWS / 8, 256, 0, stream>>>(h, ln1g, ln1b, tv, xln_b, NROWS);
    // q,k,v projections (bf16 out, feed attention directly)
    gemm_kernel<false, false, false, true><<<gD, 256, 0, stream>>>(xln_b, Wq, qbs, nullptr, nullptr, qb, NROWS, DMODEL, DMODEL);
    gemm_kernel<false, false, false, true><<<gD, 256, 0, stream>>>(xln_b, Wk, kbs, nullptr, nullptr, kb, NROWS, DMODEL, DMODEL);
    gemm_kernel<false, false, false, true><<<gD, 256, 0, stream>>>(xln_b, Wv, vbs, nullptr, nullptr, vb, NROWS, DMODEL, DMODEL);
    attn_kernel<<<8 * NHEAD * (T_TOK / 16), 32, 0, stream>>>(qb, kb, vb, ab);
    // h = h + o_proj(attn)
    gemm_kernel<false, true, false, false><<<gD, 256, 0, stream>>>(ab, Wo, obs, h, nullptr, h, NROWS, DMODEL, DMODEL);
    // tvec2; x_ = LN2(h) + tvec2
    gemm_kernel<true,  false, false, true ><<<gT, 256, 0, stream>>>(temb_b, Wt2a, t2ab, nullptr, nullptr, th_b, 8, DMODEL, DMODEL);
    gemm_kernel<false, false, false, false><<<gT, 256, 0, stream>>>(th_b,   Wt2b, t2bb, nullptr, nullptr, tv,   8, DMODEL, DMODEL);
    ln_kernel<<<NROWS / 8, 256, 0, stream>>>(h, ln2g, ln2b, tv, xln_b, NROWS);
    // h = h + f2(relu(f1(x_)))
    gemm_kernel<true,  false, false, true ><<<gF, 256, 0, stream>>>(xln_b, Wf1, f1b, nullptr, nullptr, ff_b, NROWS, DFF, DMODEL);
    gemm_kernel<false, true,  false, false><<<gD, 256, 0, stream>>>(ff_b,  Wf2, f2b, h, nullptr, h, NROWS, DMODEL, DFF);
  }

  // final LN + output head + un-patchify
  ln_kernel<<<NROWS / 8, 256, 0, stream>>>(h, fln_g, fln_b, nullptr, xln_b, NROWS);
  gemm_kernel<false, false, false, false><<<gO, 256, 0, stream>>>(xln_b, out_wt, out_b, nullptr, nullptr, outf, NROWS, PDIM, DMODEL);
  unpatchify_kernel<<<(8 * 3 * 512 * 512 + 255) / 256, 256, 0, stream>>>(outf, (float*)d_out);
  (void)in_sizes; (void)n_in; (void)out_size; (void)ws_size;
}
